// Model_386547056923
// MI455X (gfx1250) — compile-verified, matrix operand order checked by reading, not run on hardware
//
#include <hip/hip_runtime.h>
#include <math.h>

typedef __attribute__((ext_vector_type(16))) _Float16 v16h;
typedef __attribute__((ext_vector_type(8)))  _Float16 v8h;
typedef __attribute__((ext_vector_type(8)))  float    v8f;

#define NROWS 8192
#define K1    512     // n_in
#define NH    256     // n_h
#define MTILE 32      // rows per block

// Fragment from f32 memory (A-side, seq1): two contiguous 8-float runs at
// base[0..7] (K=kk+kOff..+7) and base[16..23] (K=kk+16+kOff..+7).
__device__ __forceinline__ v16h frag_from_f32(const float* __restrict__ base) {
    float4 f0 = ((const float4*)base)[0];
    float4 f1 = ((const float4*)base)[1];
    float4 f2 = ((const float4*)(base + 16))[0];
    float4 f3 = ((const float4*)(base + 16))[1];
    v16h a;
    a[0]=(_Float16)f0.x;  a[1]=(_Float16)f0.y;  a[2]=(_Float16)f0.z;  a[3]=(_Float16)f0.w;
    a[4]=(_Float16)f1.x;  a[5]=(_Float16)f1.y;  a[6]=(_Float16)f1.z;  a[7]=(_Float16)f1.w;
    a[8]=(_Float16)f2.x;  a[9]=(_Float16)f2.y;  a[10]=(_Float16)f2.z; a[11]=(_Float16)f2.w;
    a[12]=(_Float16)f3.x; a[13]=(_Float16)f3.y; a[14]=(_Float16)f3.z; a[15]=(_Float16)f3.w;
    return a;
}

// Fragment from f16 memory (pre-converted weights in ws, or LDS x1):
// two aligned 16-byte half-vectors -> b128 loads, zero conversions.
__device__ __forceinline__ v16h frag_from_h(const _Float16* __restrict__ base) {
    v8h lo = *(const v8h*)base;
    v8h hi = *(const v8h*)(base + 16);
    v16h a;
#pragma unroll
    for (int j = 0; j < 8; ++j) { a[j] = lo[j]; a[8 + j] = hi[j]; }
    return a;
}

// One-time f32 -> f16 conversion of the weight matrices into workspace.
__global__ __launch_bounds__(256) void cvt_f32_f16(
    const float* __restrict__ in, _Float16* __restrict__ out, int n)
{
    const int i = blockIdx.x * 256 + threadIdx.x;
    if (i < n) out[i] = (_Float16)in[i];
}

// One block = 256 threads (8 waves) handles 32 rows of seq1:
//   Phase 1: X1[32,256] = relu(seq1_tile @ W1^T + b1)  -> LDS (f16)
//   Phase 2: X2[32,512] = X1 @ W2^T + b2; accumulate per-row sum((seq1-X2)^2)
//   Epilogue: score[row] = sqrt(rowsum)
__global__ __launch_bounds__(256) void fused_gemm_score(
    const float*    __restrict__ seq1,  // [8192,512] f32
    const _Float16* __restrict__ W1h,   // [256,512]  f16 (pre-converted)
    const float*    __restrict__ b1,    // [256]
    const _Float16* __restrict__ W2h,   // [512,256]  f16 (pre-converted)
    const float*    __restrict__ b2,    // [512]
    float* __restrict__ score)          // [8192]
{
    __shared__ _Float16 x1[MTILE * NH];   // 16 KB
    __shared__ float rowsq[MTILE];

    const int tid   = threadIdx.x;
    const int wave  = tid >> 5;
    const int lane  = tid & 31;
    const int laneN = lane & 15;          // column / row-in-tile index
    const int kOff  = (lane >> 4) << 3;   // 0 or 8 (also row offset for C/D)
    const int r0    = blockIdx.x * MTILE;

    if (tid < MTILE) rowsq[tid] = 0.0f;

    // ---------------- Phase 1 ----------------
    {
        const int m  = wave & 1;          // 16-row subtile
        const int nb = wave >> 1;         // 0..3 -> 64-column strip
        const float* aBase = seq1 + (size_t)(r0 + m * 16 + laneN) * K1;

        v8f acc[4];
#pragma unroll
        for (int t = 0; t < 4; ++t) acc[t] = (v8f){0,0,0,0,0,0,0,0};

        for (int kk = 0; kk < K1; kk += 32) {
            v16h a = frag_from_f32(aBase + kk + kOff);
#pragma unroll
            for (int t = 0; t < 4; ++t) {
                const int bCol = nb * 64 + t * 16 + laneN;
                v16h b = frag_from_h(W1h + (size_t)bCol * K1 + kk + kOff);
                acc[t] = __builtin_amdgcn_wmma_f32_16x16x32_f16(
                    false, a, false, b, (short)0, acc[t], false, false);
            }
        }
        // bias + ReLU, restage to LDS in row-major f16 (A-fragment friendly)
#pragma unroll
        for (int t = 0; t < 4; ++t) {
            const int col  = nb * 64 + t * 16 + laneN;
            const float bb = b1[col];
#pragma unroll
            for (int g = 0; g < 8; ++g) {
                const int rowL = m * 16 + kOff + g;   // C/D layout: M = g (+8 hi lanes)
                float v = acc[t][g] + bb;
                x1[rowL * NH + col] = (_Float16)(v > 0.0f ? v : 0.0f);
            }
        }
    }
    __syncthreads();

    // ---------------- Phase 2 ----------------
    {
        const int m  = wave & 1;
        const int nb = wave >> 1;          // 0..3 -> 128-column strip of 512 outputs
        const _Float16* aBase = &x1[(m * 16 + laneN) * NH];

        v8f acc[8];
#pragma unroll
        for (int t = 0; t < 8; ++t) acc[t] = (v8f){0,0,0,0,0,0,0,0};

        for (int kk = 0; kk < NH; kk += 32) {
            v16h a = frag_from_h(aBase + kk + kOff);
#pragma unroll
            for (int t = 0; t < 8; ++t) {
                const int bCol = nb * 128 + t * 16 + laneN;
                v16h b = frag_from_h(W2h + (size_t)bCol * NH + kk + kOff);
                acc[t] = __builtin_amdgcn_wmma_f32_16x16x32_f16(
                    false, a, false, b, (short)0, acc[t], false, false);
            }
        }
        // squared-error accumulation: each (lane,g) owns a fixed row
        float sq[8] = {0,0,0,0,0,0,0,0};
#pragma unroll
        for (int t = 0; t < 8; ++t) {
            const int col  = nb * 128 + t * 16 + laneN;
            const float bb = b2[col];
#pragma unroll
            for (int g = 0; g < 8; ++g) {
                const int rowL = m * 16 + kOff + g;
                const float x2 = acc[t][g] + bb;
                const float d  = seq1[(size_t)(r0 + rowL) * K1 + col] - x2;
                sq[g] += d * d;
            }
        }
#pragma unroll
        for (int g = 0; g < 8; ++g)
            atomicAdd(&rowsq[m * 16 + kOff + g], sq[g]);  // ds_add_f32
    }
    __syncthreads();

    if (tid < MTILE) score[r0 + tid] = sqrtf(rowsq[tid]);
}

// out[0] = mean(score[idx_train]); out[1..4096] = score[idx_test]
__global__ __launch_bounds__(256) void gather_reduce(
    const float* __restrict__ score,
    const int* __restrict__ idx_train,
    const int* __restrict__ idx_test,
    float* __restrict__ out)
{
    __shared__ float red[256];
    const int tid = threadIdx.x;
    const int i   = blockIdx.x * 256 + tid;
    if (i < 4096) out[1 + i] = score[idx_test[i]];

    if (blockIdx.x == 0) {
        float s = 0.0f;
        for (int j = tid; j < 4096; j += 256) s += score[idx_train[j]];
        red[tid] = s;
        __syncthreads();
        for (int off = 128; off > 0; off >>= 1) {
            if (tid < off) red[tid] += red[tid + off];
            __syncthreads();
        }
        if (tid == 0) out[0] = red[0] * (1.0f / 4096.0f);
    }
}

extern "C" void kernel_launch(void* const* d_in, const int* in_sizes, int n_in,
                              void* d_out, int out_size, void* d_ws, size_t ws_size,
                              hipStream_t stream) {
    const float* seq1      = (const float*)d_in[0];
    // d_in[1] = adj      : dead (GCN embedding is DCE'd in the reference)
    const int*   idx_train = (const int*)d_in[2];
    const int*   idx_test  = (const int*)d_in[3];
    // d_in[4] = W_stru, d_in[5] = b_stru : dead
    const float* W1 = (const float*)d_in[6];
    const float* b1 = (const float*)d_in[7];
    const float* W2 = (const float*)d_in[8];
    const float* b2 = (const float*)d_in[9];

    float* out = (float*)d_out;         // [4097]: loss, then score_test

    // Workspace layout: score[8192] f32 | W1h[256*512] f16 | W2h[512*256] f16
    float*    score = (float*)d_ws;
    _Float16* W1h   = (_Float16*)((char*)d_ws + 32768);
    _Float16* W2h   = (_Float16*)((char*)d_ws + 32768 + (size_t)NH * K1 * 2);

    const int nW = NH * K1;  // 131072 elements each
    cvt_f32_f16<<<(nW + 255) / 256, 256, 0, stream>>>(W1, W1h, nW);
    cvt_f32_f16<<<(nW + 255) / 256, 256, 0, stream>>>(W2, W2h, nW);

    fused_gemm_score<<<NROWS / MTILE, 256, 0, stream>>>(seq1, W1h, b1, W2h, b2, score);
    gather_reduce<<<16, 256, 0, stream>>>(score, idx_train, idx_test, out);
}
